// LSTMEncoding_86328842650007
// MI455X (gfx1250) — matrix operand order repaired and checked by reading, not compiled, not run on hardware
//
#include <hip/hip_runtime.h>
#include <hip/hip_fp16.h>

typedef __attribute__((ext_vector_type(16))) _Float16 v16h;
typedef __attribute__((ext_vector_type(8)))  _Float16 v8h;
typedef __attribute__((ext_vector_type(8)))  float    v8f;
typedef unsigned int u32x4 __attribute__((ext_vector_type(4)));
typedef int          i32x4 __attribute__((ext_vector_type(4)));
typedef int          i32x8 __attribute__((ext_vector_type(8)));

constexpr int Bq = 16;
constexpr int Tq = 4096;
constexpr int Dq = 256;
constexpr int Gq = 768;            // 3*D
constexpr int SLAB = Bq * Gq;      // 12288 f16 elements = 24 KB per timestep

#if defined(__gfx1250__) && __has_builtin(__builtin_amdgcn_tensor_load_to_lds) && \
    __has_builtin(__builtin_amdgcn_s_wait_tensorcnt)
#define USE_TDM 1
#else
#define USE_TDM 0
#endif

// ---------------------------------------------------------------------------
// Kernel 0: convert W_ih / W_hh (f32) -> f16 copies in workspace
// ---------------------------------------------------------------------------
__global__ void cvt_weights(const float* __restrict__ wih,
                            const float* __restrict__ whh,
                            _Float16* __restrict__ wih16,
                            _Float16* __restrict__ whh16, int n) {
  int i = blockIdx.x * blockDim.x + threadIdx.x;
  if (i < n) {
    wih16[i] = (_Float16)wih[i];
    whh16[i] = (_Float16)whh[i];
  }
}

// ---------------------------------------------------------------------------
// Kernel 1: gx[t][b][n] = sum_k x[b,t,k] * W_ih[n,k] + b_ih[n]  (f16 out)
// Grid: 4096 M-tiles (16 x-rows). Block: 256 threads = 8 waves, 6 N-tiles/wave.
// Inner loop software-pipelined: (kc+1) fragments load under kc WMMAs.
// ---------------------------------------------------------------------------
__global__ __launch_bounds__(256) void gx_gemm(const float*    __restrict__ x,
                                               const _Float16* __restrict__ wih16,
                                               const float*    __restrict__ bih,
                                               _Float16*       __restrict__ gx) {
  const int lane = threadIdx.x & 31;
  const int wave = threadIdx.x >> 5;
  const int hi   = lane >> 4;
  const int ln   = lane & 15;
  const int m0   = blockIdx.x * 16;
  const int t0   = m0 & (Tq - 1);
  const int bidx = m0 >> 12;
  const int nbase0 = wave * 96;

  v8f acc[6];
#pragma unroll
  for (int j = 0; j < 6; ++j) {
    float bv = bih[nbase0 + j * 16 + ln];
#pragma unroll
    for (int e = 0; e < 8; ++e) acc[j][e] = bv;
  }

  const float* xrow = x + (size_t)(m0 + ln) * Dq;

  auto loadA = [&](int kc) {
    // A (16x32 f16): M = ln, K = 32*kc + 16*hi + {0..15}
    const int kb = kc * 32 + hi * 16;
    const float* p0 = xrow + kb;
    const float* p1 = xrow + kb + 8;
    v16h a;
#pragma unroll
    for (int e = 0; e < 8; ++e) {
      a[e]     = (_Float16)p0[e];
      a[8 + e] = (_Float16)p1[e];
    }
    return a;
  };
  auto loadB = [&](int kc, int j) {
    // B (32x16 f16): N = ln, K = 32*kc + 16*hi + e (contiguous 32B)
    const int n = nbase0 + j * 16 + ln;
    return *(const v16h*)(wih16 + (size_t)n * Dq + kc * 32 + hi * 16);
  };

  v16h a0 = loadA(0);
  v16h b0[6];
#pragma unroll
  for (int j = 0; j < 6; ++j) b0[j] = loadB(0, j);

#pragma unroll
  for (int kc = 0; kc < 8; ++kc) {
    v16h a1 = a0;
    v16h b1[6];
#pragma unroll
    for (int j = 0; j < 6; ++j) b1[j] = b0[j];
    if (kc < 7) {
      a1 = loadA(kc + 1);
#pragma unroll
      for (int j = 0; j < 6; ++j) b1[j] = loadB(kc + 1, j);
    }
#pragma unroll
    for (int j = 0; j < 6; ++j)
      acc[j] = __builtin_amdgcn_wmma_f32_16x16x32_f16(
          false, a0, false, b0[j], (short)0, acc[j], false, false);
    a0 = a1;
#pragma unroll
    for (int j = 0; j < 6; ++j) b0[j] = b1[j];
  }

  // C layout: VGPR i -> M = i + 8*hi, N = ln.  gx layout [t][b][n].
#pragma unroll
  for (int j = 0; j < 6; ++j) {
    const int n = nbase0 + j * 16 + ln;
#pragma unroll
    for (int i = 0; i < 8; ++i) {
      const int m = i + 8 * hi;
      gx[(size_t)((t0 + m) * Bq + bidx) * Gq + n] = (_Float16)acc[j][i];
    }
  }
}

// ---------------------------------------------------------------------------
// TDM: load one contiguous 24KB gx slab into LDS (D# per ISA 08 §8.3/8.4).
// ---------------------------------------------------------------------------
#if USE_TDM
__device__ __forceinline__ void tdm_load_slab(unsigned lds_byte_addr,
                                              const _Float16* gsrc) {
  const unsigned long long ga = (unsigned long long)(uintptr_t)gsrc;
  u32x4 g0;
  g0[0] = 1u;                                          // count=1, user D#
  g0[1] = lds_byte_addr;                               // lds_addr
  g0[2] = (unsigned)(ga & 0xFFFFFFFFu);                // global_addr lo
  g0[3] = (unsigned)((ga >> 32) & 0x01FFFFFFu)         // global_addr hi
        | (2u << 30);                                  // type = 2 (image)
  i32x8 g1;
  g1[0] = (1 << 16);                                   // data_size = 1 (2B)
  g1[1] = (int)((unsigned)SLAB << 16);                 // tensor_dim0[15:0]
  g1[2] = (1 << 16);                                   // dim0 hi=0, tensor_dim1=1
  g1[3] = (int)((unsigned)SLAB << 16);                 // tile_dim0 = 12288
  g1[4] = 1;                                           // tile_dim1=1, tile_dim2=0
  g1[5] = (int)SLAB;                                   // tensor_dim0_stride lo
  g1[6] = 0;
  g1[7] = 0;
  i32x4 gz = {0, 0, 0, 0};
#if __clang_major__ >= 23
  i32x8 gz8 = {0, 0, 0, 0, 0, 0, 0, 0};
  __builtin_amdgcn_tensor_load_to_lds(g0, g1, gz, gz, gz8, 0);
#else
  __builtin_amdgcn_tensor_load_to_lds(g0, g1, gz, gz, 0);
#endif
}
#endif

// ---------------------------------------------------------------------------
// Kernel 2: sequential GRU scan. 1 workgroup, 512 threads = 16 waves.
// Wave w owns D-columns [16w,16w+16) of all 3 gates (W_hh in 192 VGPRs/lane).
// f32 master h lives in registers (each lane owns its 8 (b,d) cells);
// f16 h copy in LDS feeds the WMMA A operand.  gx slabs double-buffered in
// LDS via TENSOR_LOAD_TO_LDS (TENSORcnt), overlapping DMA with compute.
// ---------------------------------------------------------------------------
__global__ __launch_bounds__(512) void gru_scan(const _Float16* __restrict__ gx,
                                                const _Float16* __restrict__ whh16,
                                                const float*    __restrict__ bhh,
                                                float*          __restrict__ out) {
  __shared__ _Float16 hf16[16][264];       // +8 halves pad: conflict-free reads
#if USE_TDM
  __shared__ _Float16 gbuf[2][SLAB];       // 2 x 24KB double buffer
#endif

  const int lane = threadIdx.x & 31;
  const int wave = threadIdx.x >> 5;       // 0..15
  const int hi   = lane >> 4;
  const int ln   = lane & 15;
  const int dcol = wave * 16 + ln;

  // W_hh B-fragments: N = g*256 + dcol, K = 32*kc + 16*hi + e (contiguous 32B)
  v16h bf[3][8];
#pragma unroll
  for (int g = 0; g < 3; ++g) {
    const size_t nrow = (size_t)(g * Dq + dcol) * Dq;
#pragma unroll
    for (int kc = 0; kc < 8; ++kc)
      bf[g][kc] = *(const v16h*)(whh16 + nrow + kc * 32 + hi * 16);
  }
  float bb[3];
#pragma unroll
  for (int g = 0; g < 3; ++g) bb[g] = bhh[g * Dq + dcol];

  float hreg[8];                           // f32 master h (m = i+8*hi, d = dcol)
#pragma unroll
  for (int i = 0; i < 8; ++i) hreg[i] = 0.f;

  for (int idx = threadIdx.x; idx < 16 * 264; idx += 512)
    (&hf16[0][0])[idx] = (_Float16)0.f;

#if USE_TDM
  if (wave == 0)                            // prime buffer 0 with gx[t=0]
    tdm_load_slab((unsigned)(uintptr_t)&gbuf[0][0], gx);
#endif
  __syncthreads();

  for (int t = 0; t < Tq; ++t) {
    // gh = h @ W_hh^T + b_hh  (reads hf16 only)
    v8f acc[3];
#pragma unroll
    for (int g = 0; g < 3; ++g)
#pragma unroll
      for (int e = 0; e < 8; ++e) acc[g][e] = bb[g];

#pragma unroll
    for (int kc = 0; kc < 8; ++kc) {
      const int kb = kc * 32 + hi * 16;
      const _Float16* pr = &hf16[ln][kb];
      v8h lo = *(const v8h*)(pr);
      v8h up = *(const v8h*)(pr + 8);
      v16h a;
#pragma unroll
      for (int e = 0; e < 8; ++e) { a[e] = lo[e]; a[8 + e] = up[e]; }
#pragma unroll
      for (int g = 0; g < 3; ++g)
        acc[g] = __builtin_amdgcn_wmma_f32_16x16x32_f16(
            false, a, false, bf[g][kc], (short)0, acc[g], false, false);
    }

#if USE_TDM
    if (wave == 0) __builtin_amdgcn_s_wait_tensorcnt(0);   // gbuf[t&1] done
#endif
    __syncthreads();   // hf16 reads complete; gbuf[t&1] visible to all waves
#if USE_TDM
    if (wave == 0 && t + 1 < Tq)            // kick DMA for next step's slab
      tdm_load_slab((unsigned)(uintptr_t)&gbuf[(t + 1) & 1][0],
                    gx + (size_t)(t + 1) * SLAB);
#endif

    // Gate math; C slot i holds (m = i + 8*hi, d = dcol)
#pragma unroll
    for (int i = 0; i < 8; ++i) {
      const int m = i + 8 * hi;             // batch index
#if USE_TDM
      const _Float16* slab = &gbuf[t & 1][m * Gq];
      const float xr = (float)slab[dcol];
      const float xz = (float)slab[Dq + dcol];
      const float xn = (float)slab[2 * Dq + dcol];
#else
      const size_t rowb = (size_t)(t * Bq + m) * Gq;
      const float xr = (float)gx[rowb + dcol];
      const float xz = (float)gx[rowb + Dq + dcol];
      const float xn = (float)gx[rowb + 2 * Dq + dcol];
      __builtin_prefetch(gx + rowb + SLAB, 0, 1);
#endif
      const float r = 1.f / (1.f + __expf(-(xr + acc[0][i])));
      const float z = 1.f / (1.f + __expf(-(xz + acc[1][i])));
      float ns = xn + r * acc[2][i];
      ns = fminf(fmaxf(ns, -20.f), 20.f);
      const float e2 = __expf(-2.f * ns);
      const float nt = (1.f - e2) / (1.f + e2);            // tanh
      hreg[i] = (1.f - z) * nt + z * hreg[i];
      out[(size_t)m * (Tq * Dq) + (size_t)t * Dq + dcol] = hreg[i];
    }

    // publish h_{t+1} (f16) for next step's A fragments
#pragma unroll
    for (int i = 0; i < 8; ++i) {
      const int m = i + 8 * hi;
      hf16[m][dcol] = (_Float16)hreg[i];
    }
    __syncthreads();   // h visible; all gbuf[t&1] reads done before rewrite
  }
}

// ---------------------------------------------------------------------------
extern "C" void kernel_launch(void* const* d_in, const int* in_sizes, int n_in,
                              void* d_out, int out_size, void* d_ws, size_t ws_size,
                              hipStream_t stream) {
  const float* x   = (const float*)d_in[0];
  const float* Wih = (const float*)d_in[1];
  const float* Whh = (const float*)d_in[2];
  const float* bih = (const float*)d_in[3];
  const float* bhh = (const float*)d_in[4];
  float* out = (float*)d_out;

  // Workspace: gx f16 [T][16][768] (96 MB), then f16 weight copies.
  _Float16* gx    = (_Float16*)d_ws;
  _Float16* wih16 = gx + (size_t)Tq * SLAB;
  _Float16* whh16 = wih16 + (size_t)Gq * Dq;

  cvt_weights<<<(Gq * Dq + 255) / 256, 256, 0, stream>>>(Wih, Whh, wih16, whh16, Gq * Dq);
  gx_gemm<<<(Bq * Tq) / 16, 256, 0, stream>>>(x, wih16, bih, gx);
  gru_scan<<<1, 512, 0, stream>>>(gx, whh16, bhh, out);
}
